// LlamaAttention_4561255268871
// MI455X (gfx1250) — matrix operand order repaired and checked by reading, not compiled, not run on hardware
//
#include <hip/hip_runtime.h>
#include <hip/hip_bf16.h>

typedef __bf16 bf16_t;
typedef __attribute__((ext_vector_type(16))) __bf16 v16bf;
typedef __attribute__((ext_vector_type(8)))  float  v8f;
typedef __attribute__((ext_vector_type(4)))  unsigned u32x4;
typedef __attribute__((ext_vector_type(8)))  int      i32x8;
typedef __attribute__((ext_vector_type(4)))  int      i32x4;

#define Hh        32
#define Dh        128
#define HID       4096
#define TOTAL_Q   1280
#define BSZ       64
#define NUM_BLOCKS 128
#define MAX_BLOCKS 32
#define SCALE     0.08838834764831845f   // 1/sqrt(128)

// ---------------- TDM: 2D tile (bf16) global -> LDS ----------------
// D# packing per CDNA5 ISA §8.3/8.4. All operands wave-uniform -> SGPR groups.
// 6-arg builtin variant: (g0, g1, g2, g3, g4, cpol)
__device__ __forceinline__ void tdm_load_2d_bf16(unsigned lds_off, const void* gptr,
                                                 unsigned tensor_d0, unsigned tensor_d1,
                                                 unsigned long long d0_stride_elems,
                                                 unsigned tile_d0, unsigned tile_d1) {
    unsigned long long ga = (unsigned long long)(uintptr_t)gptr;
    u32x4 g0;
    g0[0] = 1u;                                        // count=1 (valid user D#)
    g0[1] = lds_off;                                   // lds_addr [63:32]
    g0[2] = (unsigned)(ga & 0xffffffffu);              // global_addr[31:0]
    g0[3] = (unsigned)((ga >> 32) & 0x01ffffffu)       // global_addr[56:32]
          | 0x80000000u;                               // type=2 ("image") [127:126]
    i32x8 g1;
    g1[0] = (int)(1u << 16);                           // data_size=1 (2 bytes)
    g1[1] = (int)((tensor_d0 & 0xffffu) << 16);        // tensor_dim0[15:0] -> bits 63:48
    g1[2] = (int)((tensor_d0 >> 16) & 0xffffu)         // tensor_dim0[31:16]
          | (int)((tensor_d1 & 0xffffu) << 16);        // tensor_dim1[15:0]
    g1[3] = (int)((tensor_d1 >> 16) & 0xffffu)         // tensor_dim1[31:16]
          | (int)((tile_d0 & 0xffffu) << 16);          // tile_dim0 [127:112]
    g1[4] = (int)(tile_d1 & 0xffffu);                  // tile_dim1 [143:128], tile_dim2=0
    g1[5] = (int)(d0_stride_elems & 0xffffffffu);      // tensor_dim0_stride[31:0]
    g1[6] = (int)((d0_stride_elems >> 32) & 0xffffu);  // tensor_dim0_stride[47:32]
    g1[7] = 0;
    i32x4 z4 = {0, 0, 0, 0};
    i32x8 z8 = {0, 0, 0, 0, 0, 0, 0, 0};
    __builtin_amdgcn_tensor_load_to_lds(g0, g1, z4, z4, z8, 0);
}

// ---------------- fragment helpers (16x16x32 bf16 WMMA layouts) ----------------

// A-matrix 16x32 (MxK): lane<16 -> M=lane, K=0..15 ; lane>=16 -> K=16..31
__device__ __forceinline__ v16bf load_a_frag_lds(const bf16_t* p) {
    v16bf a;
#pragma unroll
    for (int i = 0; i < 8; ++i) { a[i] = p[i]; a[8 + i] = p[8 + i]; }
    return a;
}

// B-matrix 32x16 (KxN) from row-major W[N][K]: lane%16 -> N, halves split K 0-15/16-31.
__device__ __forceinline__ v16bf load_b_frag(const bf16_t* __restrict__ W, int ldb,
                                             int n0, int k0, int lane) {
    int half = lane >> 4, r = lane & 15;
    const bf16_t* p = W + (size_t)(n0 + r) * ldb + k0 + half * 16;
    v16bf b;
#pragma unroll
    for (int j = 0; j < 16; ++j) b[j] = p[j];
    return b;
}

__device__ __forceinline__ v8f zero_v8f() {
    v8f z;
#pragma unroll
    for (int i = 0; i < 8; ++i) z[i] = 0.0f;
    return z;
}

// ---------------- kernel 1: fp32 -> bf16 convert ----------------

__global__ void cvt_f32_bf16(const float* __restrict__ s, bf16_t* __restrict__ d, long n) {
    long i = (long)blockIdx.x * blockDim.x + threadIdx.x;
    long stride = (long)gridDim.x * blockDim.x;
    for (; i < n; i += stride) d[i] = (bf16_t)s[i];
}

// ---------------- kernel 2: C[M,N] = A[M,K] * W[N,K]^T  (bf16 in, f32 out) --------------
// Block = 4 waves sharing one 16-row M tile; A staged 16x256 per chunk by the TDM into
// LDS, each wave computes a 16x64 N tile streaming B from global (L2 resident).

#define KCHUNK 256

__global__ void wmma_gemm_kernel(const bf16_t* __restrict__ A,
                                 const bf16_t* __restrict__ W,
                                 float* __restrict__ C,
                                 int M, int N, int K) {
    __shared__ bf16_t Ald[16 * KCHUNK];   // 8 KB
    int lane  = threadIdx.x & 31;
    int wave  = threadIdx.x >> 5;
    int waves = blockDim.x >> 5;
    long gw   = (long)blockIdx.x * waves + wave;
    int ntile = N >> 6;                   // N/64 (multiple of 4 -> all waves share m0)
    int m0 = (int)(gw / ntile) * 16;
    int n0 = (int)(gw % ntile) * 64;
    if (m0 >= M) return;
    int half = lane >> 4, r = lane & 15;

    v8f acc[4];
#pragma unroll
    for (int t = 0; t < 4; ++t) acc[t] = zero_v8f();

    for (int kc = 0; kc < K; kc += KCHUNK) {
        if (wave == 0) {
            tdm_load_2d_bf16((unsigned)(uintptr_t)&Ald[0],
                             A + (size_t)m0 * K + kc,
                             (unsigned)(K - kc), 16u,
                             (unsigned long long)K,
                             KCHUNK, 16u);
            __builtin_amdgcn_s_wait_tensorcnt(0);
        }
        __syncthreads();                  // A chunk visible to all 4 waves
#pragma unroll
        for (int k0 = 0; k0 < KCHUNK; k0 += 32) {
            v16bf a = load_a_frag_lds(Ald + r * KCHUNK + k0 + half * 16);
            __builtin_prefetch(W + (size_t)(n0 + r) * K + kc + k0 + 32, 0, 1);
#pragma unroll
            for (int t = 0; t < 4; ++t) {
                v16bf b = load_b_frag(W, K, n0 + 16 * t, kc + k0, lane);
                acc[t] = __builtin_amdgcn_wmma_f32_16x16x32_bf16(
                    false, a, false, b, (short)0, acc[t], false, false);
            }
        }
        __syncthreads();                  // safe to overwrite Ald
    }
#pragma unroll
    for (int t = 0; t < 4; ++t)
#pragma unroll
        for (int i = 0; i < 8; ++i)
            C[(size_t)(m0 + half * 8 + i) * N + n0 + 16 * t + r] = acc[t][i];
}

// ---------------- kernel 3: RoPE q,k + scatter k,v into paged bf16 cache ----------------

__global__ void rope_scatter_kernel(const float* __restrict__ qf,
                                    const float* __restrict__ kf,
                                    const float* __restrict__ vf,
                                    const int* __restrict__ position_ids,
                                    const int* __restrict__ block_offsets,
                                    bf16_t* __restrict__ qb,
                                    bf16_t* __restrict__ kcache,
                                    bf16_t* __restrict__ vcache) {
    int idx = blockIdx.x * blockDim.x + threadIdx.x;
    if (idx >= TOTAL_Q * Hh * 64) return;
    int j = idx & 63;
    int h = (idx >> 6) & 31;
    int t = idx >> 11;
    int pos = position_ids[t];
    int seq = (t < 256) ? 0 : (t < 768) ? 1 : (t < 896) ? 2 : 3;

    float inv = __expf((float)(-2 * j) * (1.0f / 128.0f) * 9.210340371976184f); // ln(1e4)
    float ang = (float)pos * inv;
    float c = __cosf(ang), s = __sinf(ang);

    size_t base = ((size_t)t * Hh + h) * Dh;
    float q1 = qf[base + j], q2 = qf[base + j + 64];
    qb[base + j]      = (bf16_t)(q1 * c - q2 * s);
    qb[base + j + 64] = (bf16_t)(q2 * c + q1 * s);

    int dest = block_offsets[seq * MAX_BLOCKS + (pos >> 6)] * BSZ + (pos & 63);
    size_t cb = ((size_t)dest * Hh + h) * Dh;
    float k1 = kf[base + j], k2 = kf[base + j + 64];
    kcache[cb + j]      = (bf16_t)(k1 * c - k2 * s);
    kcache[cb + j + 64] = (bf16_t)(k2 * c + k1 * s);
    vcache[cb + j]      = (bf16_t)vf[base + j];
    vcache[cb + j + 64] = (bf16_t)vf[base + j + 64];
}

// ---------------- kernel 4: flash attention, 1 wave per (seq, head, 16-row q tile) -------
// K and V 32x128 chunks staged by the TDM into LDS; all operand reads become DS ops.

__global__ void attn_kernel(const bf16_t* __restrict__ qb,
                            const bf16_t* __restrict__ kcache,
                            const bf16_t* __restrict__ vcache,
                            const int* __restrict__ block_offsets,
                            bf16_t* __restrict__ ob) {
    __shared__ bf16_t KL[32 * Dh];    // 8 KB
    __shared__ bf16_t VL[32 * Dh];    // 8 KB
    __shared__ bf16_t Pld[16 * 32];   // 1 KB
    int lane = threadIdx.x & 31;
    int half = lane >> 4, r = lane & 15;
    int gt = blockIdx.x;              // 0..79 global q tile
    int h  = blockIdx.y;

    int b, qt;
    if      (gt < 16) { b = 0; qt = gt; }
    else if (gt < 48) { b = 1; qt = gt - 16; }
    else if (gt < 56) { b = 2; qt = gt - 48; }
    else              { b = 3; qt = gt - 56; }
    const int hist[4]   = {1792, 1024, 1536, 768};
    const int qstart[4] = {0, 256, 768, 896};
    int pos0 = hist[b] + qt * 16;
    int tok0 = qstart[b] + qt * 16;
    const int* bt = block_offsets + b * MAX_BLOCKS;

    // Q fragments: A-matrix 16 x 128 (4 K-chunks of 32)
    v16bf qa[4];
#pragma unroll
    for (int kk = 0; kk < 4; ++kk) {
        const bf16_t* p = qb + (size_t)(tok0 + r) * HID + h * Dh + kk * 32 + half * 16;
        qa[kk] = load_a_frag_lds(p);  // same 8+8 contiguous pattern works on global
    }

    v8f o[8];
    float m[8], l[8];
#pragma unroll
    for (int t = 0; t < 8; ++t) o[t] = zero_v8f();
#pragma unroll
    for (int i = 0; i < 8; ++i) { m[i] = -3.0e38f; l[i] = 0.0f; }

    int nkv = pos0 + 16;                      // causal kv extent for this tile
    for (int kv0 = 0; kv0 < nkv; kv0 += 32) {
        // 32-wide chunk never crosses a 64-token page -> one table lookup
        int vbase = bt[kv0 >> 6] * BSZ + (kv0 & 63);

        // TDM: stage K and V chunks (32 rows x 128, row stride H*D) into LDS
        tdm_load_2d_bf16((unsigned)(uintptr_t)&KL[0],
                         kcache + ((size_t)vbase * Hh + h) * Dh,
                         (unsigned)Dh, 32u, (unsigned long long)(Hh * Dh), Dh, 32u);
        tdm_load_2d_bf16((unsigned)(uintptr_t)&VL[0],
                         vcache + ((size_t)vbase * Hh + h) * Dh,
                         (unsigned)Dh, 32u, (unsigned long long)(Hh * Dh), Dh, 32u);
        __builtin_amdgcn_s_wait_tensorcnt(0);

        v8f s0 = zero_v8f(), s1 = zero_v8f();
#pragma unroll
        for (int kk = 0; kk < 4; ++kk) {
            // B-frag for S = Q x K^T: lane r -> kv col; element j -> d
            const bf16_t* p0 = KL + (size_t)r * Dh + kk * 32 + half * 16;
            const bf16_t* p1 = KL + (size_t)(16 + r) * Dh + kk * 32 + half * 16;
            v16bf b0, b1;
#pragma unroll
            for (int j = 0; j < 16; ++j) { b0[j] = p0[j]; b1[j] = p1[j]; }
            s0 = __builtin_amdgcn_wmma_f32_16x16x32_bf16(false, qa[kk], false, b0,
                                                         (short)0, s0, false, false);
            s1 = __builtin_amdgcn_wmma_f32_16x16x32_bf16(false, qa[kk], false, b1,
                                                         (short)0, s1, false, false);
        }

        // online softmax over the 32 columns of this chunk
        float alpha[8];
#pragma unroll
        for (int i = 0; i < 8; ++i) {
            int prow = pos0 + half * 8 + i;
            float x0 = (kv0 + r      <= prow) ? s0[i] * SCALE : -3.0e38f;
            float x1 = (kv0 + 16 + r <= prow) ? s1[i] * SCALE : -3.0e38f;
            float mx = fmaxf(x0, x1);
            mx = fmaxf(mx, __shfl_xor(mx, 1, 32));
            mx = fmaxf(mx, __shfl_xor(mx, 2, 32));
            mx = fmaxf(mx, __shfl_xor(mx, 4, 32));
            mx = fmaxf(mx, __shfl_xor(mx, 8, 32));
            float mn = fmaxf(m[i], mx);
            alpha[i] = __expf(m[i] - mn);
            float e0 = __expf(x0 - mn);
            float e1 = __expf(x1 - mn);
            float rs = e0 + e1;
            rs += __shfl_xor(rs, 1, 32);
            rs += __shfl_xor(rs, 2, 32);
            rs += __shfl_xor(rs, 4, 32);
            rs += __shfl_xor(rs, 8, 32);
            l[i] = l[i] * alpha[i] + rs;
            m[i] = mn;
            s0[i] = e0; s1[i] = e1;
        }

        // P (C-layout) -> LDS -> A-layout fragment
#pragma unroll
        for (int i = 0; i < 8; ++i) {
            Pld[(half * 8 + i) * 32 + r]      = (bf16_t)s0[i];
            Pld[(half * 8 + i) * 32 + 16 + r] = (bf16_t)s1[i];
        }
        __syncthreads();
        v16bf pa;
        {
            const bf16_t* p = Pld + r * 32 + half * 16;
#pragma unroll
            for (int i = 0; i < 8; ++i) { pa[i] = p[i]; pa[8 + i] = p[8 + i]; }
        }
        __syncthreads();

        // rescale O, then O += P x V   (V chunk from LDS as 8 B-fragments)
#pragma unroll
        for (int t = 0; t < 8; ++t) {
#pragma unroll
            for (int i = 0; i < 8; ++i) o[t][i] *= alpha[i];
            const bf16_t* vp = VL + (size_t)(half * 16) * Dh + t * 16 + r;
            v16bf vb;
#pragma unroll
            for (int j = 0; j < 16; ++j) vb[j] = vp[(size_t)j * Dh];
            o[t] = __builtin_amdgcn_wmma_f32_16x16x32_bf16(false, pa, false, vb,
                                                           (short)0, o[t], false, false);
        }
        __syncthreads();                   // KL/VL reusable next chunk
    }

    float inv_l[8];
#pragma unroll
    for (int i = 0; i < 8; ++i) inv_l[i] = 1.0f / l[i];
#pragma unroll
    for (int t = 0; t < 8; ++t)
#pragma unroll
        for (int i = 0; i < 8; ++i)
            ob[(size_t)(tok0 + half * 8 + i) * HID + h * Dh + t * 16 + r] =
                (bf16_t)(o[t][i] * inv_l[i]);
}

// ---------------- host launch ----------------

extern "C" void kernel_launch(void* const* d_in, const int* in_sizes, int n_in,
                              void* d_out, int out_size, void* d_ws, size_t ws_size,
                              hipStream_t stream) {
    (void)in_sizes; (void)n_in; (void)out_size; (void)ws_size;
    const float* hs  = (const float*)d_in[0];
    const float* wq  = (const float*)d_in[1];
    const float* wk  = (const float*)d_in[2];
    const float* wv  = (const float*)d_in[3];
    const float* wo  = (const float*)d_in[4];
    const float* pk  = (const float*)d_in[5];
    const float* pv  = (const float*)d_in[6];
    const int*   pos = (const int*)d_in[7];
    const int*   bo  = (const int*)d_in[8];

    const long HB    = (long)TOTAL_Q * HID;              // 5,242,880
    const long WB    = (long)HID * HID;                  // 16,777,216
    const long CACHE = (long)NUM_BLOCKS * BSZ * Hh * Dh; // 33,554,432

    size_t off = 0;
    auto alloc = [&](size_t bytes) {
        char* p = (char*)d_ws + off;
        off += (bytes + 255) & ~(size_t)255;
        return p;
    };
    bf16_t* hb     = (bf16_t*)alloc(HB * 2);
    bf16_t* wqb    = (bf16_t*)alloc(WB * 2);
    bf16_t* wkb    = (bf16_t*)alloc(WB * 2);
    bf16_t* wvb    = (bf16_t*)alloc(WB * 2);
    bf16_t* wob    = (bf16_t*)alloc(WB * 2);
    bf16_t* kcache = (bf16_t*)alloc(CACHE * 2);
    bf16_t* vcache = (bf16_t*)alloc(CACHE * 2);
    float*  qf     = (float*)alloc(HB * 4);
    float*  kf     = (float*)alloc(HB * 4);
    float*  vf     = (float*)alloc(HB * 4);
    bf16_t* qbf    = (bf16_t*)alloc(HB * 2);
    bf16_t* attnb  = (bf16_t*)alloc(HB * 2);

    // 1) converts
    cvt_f32_bf16<<<2048, 256, 0, stream>>>(hs, hb, HB);
    cvt_f32_bf16<<<2048, 256, 0, stream>>>(wq, wqb, WB);
    cvt_f32_bf16<<<2048, 256, 0, stream>>>(wk, wkb, WB);
    cvt_f32_bf16<<<2048, 256, 0, stream>>>(wv, wvb, WB);
    cvt_f32_bf16<<<2048, 256, 0, stream>>>(wo, wob, WB);
    cvt_f32_bf16<<<2048, 256, 0, stream>>>(pk, kcache, CACHE);
    cvt_f32_bf16<<<2048, 256, 0, stream>>>(pv, vcache, CACHE);

    // 2) QKV projections: (TOTAL_Q/16)*(HID/64) waves = 5120 / 4 per block
    int gemm_blocks = (TOTAL_Q / 16) * (HID / 64) / 4;   // 1280
    wmma_gemm_kernel<<<gemm_blocks, 128, 0, stream>>>(hb, wqb, qf, TOTAL_Q, HID, HID);
    wmma_gemm_kernel<<<gemm_blocks, 128, 0, stream>>>(hb, wkb, kf, TOTAL_Q, HID, HID);
    wmma_gemm_kernel<<<gemm_blocks, 128, 0, stream>>>(hb, wvb, vf, TOTAL_Q, HID, HID);

    // 3) RoPE + cache scatter
    int rope_threads = TOTAL_Q * Hh * 64;
    rope_scatter_kernel<<<(rope_threads + 255) / 256, 256, 0, stream>>>(
        qf, kf, vf, pos, bo, qbf, kcache, vcache);

    // 4) attention: 80 q-tiles x 32 heads, one wave each
    attn_kernel<<<dim3(80, 32), 32, 0, stream>>>(qbf, kcache, vcache, bo, attnb);

    // 5) output projection -> fp32 d_out
    wmma_gemm_kernel<<<gemm_blocks, 128, 0, stream>>>(attnb, wob, (float*)d_out,
                                                      TOTAL_Q, HID, HID);
}
